// TransformerBlock_64690797412809
// MI455X (gfx1250) — matrix operand order repaired and checked by reading, not compiled, not run on hardware
//
#include <hip/hip_runtime.h>

// ---------------------------------------------------------------------------
// CDNA5 (gfx1250) transformer block forward.
// B=2, S=1024, D=2048, H=16, KVH=4, HD=128, HID=4096, E=8, K=2
// All GEMMs via v_wmma_f32_16x16x32_bf16 (bf16 in, fp32 accum).
// All B operands pre-transposed to N x K bf16 (one-time pass), so GEMM
// staging is pure b128 traffic. Block tile 64x128, 4 waves, wave tile 32x64
// (2x4 WMMA tiles), double-buffered LDS, 1 barrier per K-step.
// ---------------------------------------------------------------------------

typedef __attribute__((ext_vector_type(16))) __bf16 v16bf;
typedef __attribute__((ext_vector_type(8)))  __bf16 v8bf;
typedef __attribute__((ext_vector_type(8)))  float  v8f;
typedef long long i64;

// ---------------------------------------------------------------------------
// Generic tiled bf16 WMMA GEMM:  C[z] = A[z] @ B[z]^T-layout  (+res | scatter)
//   - B is ALWAYS stored N x Kd row-major (pre-transposed weights / K / vT)
//   - batched via z: zb = z/zdiv, zh = z%zdiv, B head index = zh/bDiv
//   - gatherA: row gather list for A (MoE token lists), hoisted out of K loop
//   - scatterC/scaleC: C[scatter[m]] += scale[m]*acc (MoE expert output)
//   - mCount: dynamic row count (blocks past it early-exit)
// ---------------------------------------------------------------------------
__global__ __launch_bounds__(128)
void gemm_bf16_wmma(const __bf16* __restrict__ A, const __bf16* __restrict__ Bm,
                    float* __restrict__ C,
                    int M, int N, int Kd, int lda, int ldb, int ldc,
                    int zdiv, int bDiv,
                    i64 saB, i64 saH, i64 sbB, i64 sbH, i64 scB, i64 scH,
                    const float* __restrict__ residual, int ldr,
                    const int* __restrict__ gatherA,
                    const int* __restrict__ scatterC,
                    const float* __restrict__ scaleC,
                    const int* __restrict__ mCount)
{
    __shared__ __bf16 As[2][64][40];    // M x K tiles (ping-pong)
    __shared__ __bf16 Bs[2][128][40];   // N x K tiles

    const int tid = threadIdx.x;
    const int z  = blockIdx.z;
    const int zb = z / zdiv, zh = z % zdiv;
    A  += zb * saB + (i64)zh * saH;
    Bm += zb * sbB + (i64)(zh / bDiv) * sbH;
    C  += zb * scB + (i64)zh * scH;

    const int mBase = blockIdx.y * 64;
    const int nBase = blockIdx.x * 128;
    int Meff = M;
    if (mCount) { Meff = *mCount; if (mBase >= Meff) return; }

    const int wave = tid >> 5, lane = tid & 31;
    const int wm = (wave >> 1) * 32, wn = (wave & 1) * 64;
    const int hl = lane >> 4, l15 = lane & 15;

    // ---- loop-invariant staging pointers (gather hoisted out of K loop) ----
    const int rA = tid >> 1, segA = tid & 1;      // A: 64 rows x 2 chunks
    {
        int g = mBase + rA;
        int src = g;
        if (gatherA) src = gatherA[(g < Meff) ? g : (Meff - 1)];
        A += (i64)src * lda + segA * 16;          // this thread's chunk at k=0
    }
    const __bf16* bPtr = Bm + (i64)(nBase + tid) * ldb;   // 128 n-rows, 32 k each

    v8f acc[2][4];
#pragma unroll
    for (int a = 0; a < 2; ++a)
#pragma unroll
        for (int b = 0; b < 4; ++b) acc[a][b] = (v8f)(0.0f);

    const int nK = Kd >> 5;

    // ---- prologue: stage tile 0 into buffer 0 ----
    {
        v8bf a0 = *(const v8bf*)(A);
        v8bf a1 = *(const v8bf*)(A + 8);
        v8bf b0 = *(const v8bf*)(bPtr);
        v8bf b1 = *(const v8bf*)(bPtr + 8);
        v8bf b2 = *(const v8bf*)(bPtr + 16);
        v8bf b3 = *(const v8bf*)(bPtr + 24);
        *(v8bf*)&As[0][rA][segA * 16]     = a0;
        *(v8bf*)&As[0][rA][segA * 16 + 8] = a1;
        *(v8bf*)&Bs[0][tid][0]  = b0;
        *(v8bf*)&Bs[0][tid][8]  = b1;
        *(v8bf*)&Bs[0][tid][16] = b2;
        *(v8bf*)&Bs[0][tid][24] = b3;
    }
    __syncthreads();

    int p = 0;
    for (int kt = 0; kt < nK; ++kt) {
        const bool more = (kt + 1) < nK;
        const int k0n = (kt + 1) << 5;

        // ---- issue next tile's global loads into registers ----
        v8bf a0, a1, b0, b1, b2, b3;
        if (more) {
            const __bf16* ga = A + k0n;
            const __bf16* gb = bPtr + k0n;
            a0 = *(const v8bf*)(ga);
            a1 = *(const v8bf*)(ga + 8);
            b0 = *(const v8bf*)(gb);
            b1 = *(const v8bf*)(gb + 8);
            b2 = *(const v8bf*)(gb + 16);
            b3 = *(const v8bf*)(gb + 24);
            __builtin_prefetch(ga + 64, 0, 1);
            __builtin_prefetch(gb + 64, 0, 1);
        }

        // ---- compute from buffer p (documented 16-bit A/B lane layouts) ----
        union V16 { v16bf v; v8bf h[2]; };
        v16bf af[2], bfrag[4];
#pragma unroll
        for (int tm = 0; tm < 2; ++tm) {
            const __bf16* ar = &As[p][wm + tm * 16 + l15][0];
            V16 u;
            u.h[0] = *(const v8bf*)(ar + 8 * hl);        // K = 8*hl .. +7
            u.h[1] = *(const v8bf*)(ar + 16 + 8 * hl);   // K = 16+8*hl .. +7
            af[tm] = u.v;
        }
#pragma unroll
        for (int tn = 0; tn < 4; ++tn) {
            const __bf16* br = &Bs[p][wn + tn * 16 + l15][0];
            V16 u;
            u.h[0] = *(const v8bf*)(br + 8 * hl);        // K = 8*hl .. +7
            u.h[1] = *(const v8bf*)(br + 16 + 8 * hl);   // K = 16+8*hl .. +7
            bfrag[tn] = u.v;
        }
#pragma unroll
        for (int tm = 0; tm < 2; ++tm)
#pragma unroll
            for (int tn = 0; tn < 4; ++tn)
                acc[tm][tn] = __builtin_amdgcn_wmma_f32_16x16x32_bf16(
                    false, af[tm], false, bfrag[tn], (short)0, acc[tm][tn],
                    false, false);

        // ---- store next tile into the other buffer ----
        if (more) {
            const int q = p ^ 1;
            *(v8bf*)&As[q][rA][segA * 16]     = a0;
            *(v8bf*)&As[q][rA][segA * 16 + 8] = a1;
            *(v8bf*)&Bs[q][tid][0]  = b0;
            *(v8bf*)&Bs[q][tid][8]  = b1;
            *(v8bf*)&Bs[q][tid][16] = b2;
            *(v8bf*)&Bs[q][tid][24] = b3;
        }
        __syncthreads();
        p ^= 1;
    }

    // ---- epilogue: C/D layout: col = l15, row = r + 8*hl ----
#pragma unroll
    for (int tm = 0; tm < 2; ++tm) {
#pragma unroll
        for (int tn = 0; tn < 4; ++tn) {
            const int gc = nBase + wn + tn * 16 + l15;
#pragma unroll
            for (int r = 0; r < 8; ++r) {
                const int gr = mBase + wm + tm * 16 + r + 8 * hl;
                float v = acc[tm][tn][r];
                if (scatterC) {
                    if (gr < Meff) {
                        int tok = scatterC[gr];
                        C[(i64)tok * ldc + gc] += scaleC[gr] * v;
                    }
                } else {
                    if (residual) v += residual[(i64)gr * ldr + gc];
                    C[(i64)gr * ldc + gc] = v;
                }
            }
        }
    }
}

// ---------------------------------------------------------------------------
// Tiled transpose + fp32->bf16:  out[c][r] = (bf16) in[r][c]
// in: R x Cn fp32 (batched), out: Cn x R bf16 (batched)
// ---------------------------------------------------------------------------
__global__ __launch_bounds__(256)
void transpose_cvt_bf16(const float* __restrict__ in, __bf16* __restrict__ out,
                        int R, int Cn, i64 inBS, i64 outBS)
{
    __shared__ float tile[32][33];
    in  += (i64)blockIdx.z * inBS;
    out += (i64)blockIdx.z * outBS;
    const int c0 = blockIdx.x * 32;
    const int r0 = blockIdx.y * 32;
    const int tx = threadIdx.x & 31;
    const int ty = threadIdx.x >> 5;          // 0..7
#pragma unroll
    for (int i = 0; i < 32; i += 8)
        tile[ty + i][tx] = in[(i64)(r0 + ty + i) * Cn + c0 + tx];
    __syncthreads();
#pragma unroll
    for (int i = 0; i < 32; i += 8)
        out[(i64)(c0 + ty + i) * R + r0 + tx] = (__bf16)tile[tx][ty + i];
}

// ---------------------------------------------------------------------------
// RMSNorm (fp32 in, bf16 out): one block per row
// ---------------------------------------------------------------------------
__global__ __launch_bounds__(256)
void rmsnorm_bf16(const float* __restrict__ x, const float* __restrict__ w,
                  __bf16* __restrict__ out, int D_)
{
    const i64 row = blockIdx.x;
    const float* xr = x + row * D_;
    float ss = 0.f;
    for (int d = threadIdx.x; d < D_; d += 256) { float v = xr[d]; ss += v * v; }
    __shared__ float red[256];
    red[threadIdx.x] = ss; __syncthreads();
    for (int s = 128; s > 0; s >>= 1) {
        if (threadIdx.x < s) red[threadIdx.x] += red[threadIdx.x + s];
        __syncthreads();
    }
    const float rs = rsqrtf(red[0] / (float)D_ + 1e-5f);
    for (int d = threadIdx.x; d < D_; d += 256)
        out[row * D_ + d] = (__bf16)(xr[d] * rs * w[d]);
}

// ---------------------------------------------------------------------------
// RoPE + fp32->bf16 (layout [tok, nH, 128], pairs interleaved)
// ---------------------------------------------------------------------------
__global__ void rope_cvt(const float* __restrict__ qin, __bf16* __restrict__ qout,
                         const float* __restrict__ cosT, const float* __restrict__ sinT,
                         int S_, int nH, i64 total)
{
    i64 i = (i64)blockIdx.x * blockDim.x + threadIdx.x;
    if (i >= total) return;                    // total = T * nH * 64
    const int p = (int)(i % 64);
    i64 t = i / 64;
    const int h = (int)(t % nH);
    const i64 tok = t / nH;
    const int s = (int)(tok % S_);
    const float* base = qin + (tok * nH + h) * 128;
    const float xr = base[2 * p], xi = base[2 * p + 1];
    const float c = cosT[s * 64 + p], sn = sinT[s * 64 + p];
    __bf16* ob = qout + (tok * nH + h) * 128;
    ob[2 * p]     = (__bf16)(xr * c - xi * sn);
    ob[2 * p + 1] = (__bf16)(xr * sn + xi * c);
}

// ---------------------------------------------------------------------------
// Causal softmax over scores rows [B*H*S, S]; bf16 probs out; k>q -> 0
// ---------------------------------------------------------------------------
__global__ __launch_bounds__(128)
void softmax_causal(const float* __restrict__ scores, __bf16* __restrict__ probs,
                    int S_, float scale)
{
    const i64 row = blockIdx.x;
    const int q = (int)(row % S_);
    const float* sr = scores + row * S_;
    __bf16* pr = probs + row * S_;
    const int L = q + 1;
    __shared__ float red[128];

    float mx = -3.0e38f;
    for (int k = threadIdx.x; k < L; k += 128) mx = fmaxf(mx, sr[k] * scale);
    red[threadIdx.x] = mx; __syncthreads();
    for (int s = 64; s > 0; s >>= 1) {
        if (threadIdx.x < s) red[threadIdx.x] = fmaxf(red[threadIdx.x], red[threadIdx.x + s]);
        __syncthreads();
    }
    mx = red[0]; __syncthreads();

    float sum = 0.f;
    for (int k = threadIdx.x; k < L; k += 128) sum += __expf(sr[k] * scale - mx);
    red[threadIdx.x] = sum; __syncthreads();
    for (int s = 64; s > 0; s >>= 1) {
        if (threadIdx.x < s) red[threadIdx.x] += red[threadIdx.x + s];
        __syncthreads();
    }
    const float inv = 1.0f / red[0];

    for (int k = threadIdx.x; k < S_; k += 128) {
        float v = (k <= q) ? __expf(sr[k] * scale - mx) * inv : 0.0f;
        pr[k] = (__bf16)v;
    }
}

// ---------------------------------------------------------------------------
// MoE gate: logits = t @ gate_w, top-2 softmax, build per-expert lists
// ---------------------------------------------------------------------------
__global__ __launch_bounds__(128)
void gate_topk(const __bf16* __restrict__ t, const float* __restrict__ gw,
               float* __restrict__ coefL, int* __restrict__ idxL,
               int* __restrict__ counts, int T_, int D_)
{
    const int row = blockIdx.x;
    __shared__ float red[128 * 8];
    float p[8];
#pragma unroll
    for (int e = 0; e < 8; ++e) p[e] = 0.f;
    for (int d = threadIdx.x; d < D_; d += 128) {
        const float tv = (float)t[(i64)row * D_ + d];
#pragma unroll
        for (int e = 0; e < 8; ++e) p[e] += tv * gw[d * 8 + e];
    }
#pragma unroll
    for (int e = 0; e < 8; ++e) red[threadIdx.x * 8 + e] = p[e];
    __syncthreads();
    for (int s = 64; s > 0; s >>= 1) {
        if (threadIdx.x < s)
#pragma unroll
            for (int e = 0; e < 8; ++e)
                red[threadIdx.x * 8 + e] += red[(threadIdx.x + s) * 8 + e];
        __syncthreads();
    }
    if (threadIdx.x == 0) {
        float l[8];
#pragma unroll
        for (int e = 0; e < 8; ++e) l[e] = red[e];
        int i0 = 0;
        for (int e = 1; e < 8; ++e) if (l[e] > l[i0]) i0 = e;
        int i1 = (i0 == 0) ? 1 : 0;
        for (int e = 0; e < 8; ++e) if (e != i0 && l[e] > l[i1]) i1 = e;
        const float e1 = __expf(l[i1] - l[i0]);
        const float inv = 1.0f / (1.0f + e1);
        const float w0 = inv, w1 = e1 * inv;
        int s0 = atomicAdd(&counts[i0], 1);
        idxL[i0 * T_ + s0] = row; coefL[i0 * T_ + s0] = w0;
        int s1 = atomicAdd(&counts[i1], 1);
        idxL[i1 * T_ + s1] = row; coefL[i1 * T_ + s1] = w1;
    }
}

// ---------------------------------------------------------------------------
// elementwise helpers
// ---------------------------------------------------------------------------
__global__ void cvt_f32_bf16(const float* __restrict__ in, __bf16* __restrict__ out, i64 n)
{
    i64 i = (i64)blockIdx.x * blockDim.x + threadIdx.x;
    const i64 stride = (i64)gridDim.x * blockDim.x;
    for (; i < n; i += stride) out[i] = (__bf16)in[i];
}

__global__ void copy_f32(const float* __restrict__ in, float* __restrict__ out, i64 n)
{
    i64 i = (i64)blockIdx.x * blockDim.x + threadIdx.x;
    const i64 stride = (i64)gridDim.x * blockDim.x;
    for (; i < n; i += stride) out[i] = in[i];
}

__global__ void zero_i32(int* p, int n)
{
    int i = blockIdx.x * blockDim.x + threadIdx.x;
    if (i < n) p[i] = 0;
}

__global__ void silu_mul_bf16(const float* __restrict__ g, const float* __restrict__ u,
                              __bf16* __restrict__ out, i64 n)
{
    i64 i = (i64)blockIdx.x * blockDim.x + threadIdx.x;
    const i64 stride = (i64)gridDim.x * blockDim.x;
    for (; i < n; i += stride) {
        const float gv = g[i];
        const float s = gv / (1.0f + __expf(-gv));
        out[i] = (__bf16)(s * u[i]);
    }
}

// ---------------------------------------------------------------------------
// host orchestration
// ---------------------------------------------------------------------------
extern "C" void kernel_launch(void* const* d_in, const int* in_sizes, int n_in,
                              void* d_out, int out_size, void* d_ws, size_t ws_size,
                              hipStream_t stream)
{
    (void)in_sizes; (void)n_in; (void)out_size; (void)ws_size;

    const i64 Bn = 2, S = 1024, D = 2048, H = 16, KVH = 4, HD = 128;
    const i64 HID = 4096, E = 8;
    const i64 T = Bn * S;               // 2048 tokens
    const i64 Qd = H * HD;              // 2048
    const i64 KVd = KVH * HD;           // 512
    const float scale = 0.08838834764831845f;  // 1/sqrt(128)

    const float* x    = (const float*)d_in[0];
    const float* fcos = (const float*)d_in[1];
    const float* fsin = (const float*)d_in[2];
    // d_in[3] positions, d_in[4] mask: causal handled analytically
    const float* anw  = (const float*)d_in[5];
    const float* fnw  = (const float*)d_in[6];
    const float* wq   = (const float*)d_in[7];
    const float* wk   = (const float*)d_in[8];
    const float* wv   = (const float*)d_in[9];
    const float* wo   = (const float*)d_in[10];
    const float* gw   = (const float*)d_in[11];
    const float* w1   = (const float*)d_in[12];
    const float* w2   = (const float*)d_in[13];
    const float* w3   = (const float*)d_in[14];
    float* out = (float*)d_out;

    // ---- workspace layout (all weight matrices stored transposed: N x K) ---
    char* base = (char*)d_ws;
    size_t off = 0;
    auto alloc = [&](size_t bytes) -> void* {
        void* p = base + off;
        off = (off + bytes + 255) & ~(size_t)255;
        return p;
    };
    __bf16* wqT = (__bf16*)alloc(Qd * D * 2);        // [Qd][D]
    __bf16* wkT = (__bf16*)alloc(KVd * D * 2);       // [KVd][D]
    __bf16* wvT = (__bf16*)alloc(KVd * D * 2);       // [KVd][D]
    __bf16* woT = (__bf16*)alloc(D * Qd * 2);        // [D][Qd]
    __bf16* w1T = (__bf16*)alloc(E * HID * D * 2);   // [E][HID][D]
    __bf16* w2T = (__bf16*)alloc(E * D * HID * 2);   // [E][D][HID]
    __bf16* w3T = (__bf16*)alloc(E * HID * D * 2);   // [E][HID][D]
    __bf16* h_b  = (__bf16*)alloc(T * D * 2);
    float*  qf   = (float*)alloc(T * Qd * 4);
    float*  kf   = (float*)alloc(T * KVd * 4);
    float*  vf   = (float*)alloc(T * KVd * 4);
    __bf16* q_b  = (__bf16*)alloc(T * Qd * 2);
    __bf16* k_b  = (__bf16*)alloc(T * KVd * 2);
    __bf16* vT_b = (__bf16*)alloc(Bn * KVd * S * 2); // [b][kvh*hd][s]
    float*  attn_f = (float*)alloc(T * Qd * 4);
    __bf16* attn_b = (__bf16*)alloc(T * Qd * 2);
    float*  h1   = (float*)alloc(T * D * 4);
    __bf16* t_b  = (__bf16*)alloc(T * D * 2);
    int*    counts = (int*)alloc(E * 4);
    int*    idxL   = (int*)alloc(E * T * 4);
    float*  coefL  = (float*)alloc(E * T * 4);
    // big region: scores+probs (attention), later aliased by MoE scratch
    size_t regionOff = off;
    float*  scores = (float*)alloc(Bn * H * S * S * 4);    // 134 MB
    __bf16* probs  = (__bf16*)alloc(Bn * H * S * S * 2);   // 67 MB
    // MoE scratch aliases the (then-dead) attention region
    float*  gbuf = (float*)(base + regionOff);
    float*  ubuf = gbuf + T * HID;
    __bf16* a_b  = (__bf16*)(ubuf + T * HID);
    (void)scores;

    const dim3 thr128(128), thr256(256);

    // 1) weights fp32 -> bf16 with transpose to N x K
    transpose_cvt_bf16<<<dim3(64, 64, 1), thr256, 0, stream>>>(wq, wqT, (int)D, (int)Qd, 0, 0);
    transpose_cvt_bf16<<<dim3(16, 64, 1), thr256, 0, stream>>>(wk, wkT, (int)D, (int)KVd, 0, 0);
    transpose_cvt_bf16<<<dim3(16, 64, 1), thr256, 0, stream>>>(wv, wvT, (int)D, (int)KVd, 0, 0);
    transpose_cvt_bf16<<<dim3(64, 64, 1), thr256, 0, stream>>>(wo, woT, (int)Qd, (int)D, 0, 0);
    transpose_cvt_bf16<<<dim3(128, 64, (unsigned)E), thr256, 0, stream>>>(
        w1, w1T, (int)D, (int)HID, D * HID, HID * D);
    transpose_cvt_bf16<<<dim3(64, 128, (unsigned)E), thr256, 0, stream>>>(
        w2, w2T, (int)HID, (int)D, HID * D, D * HID);
    transpose_cvt_bf16<<<dim3(128, 64, (unsigned)E), thr256, 0, stream>>>(
        w3, w3T, (int)D, (int)HID, D * HID, HID * D);

    // 2) h = rmsnorm(x, attn_norm_w)
    rmsnorm_bf16<<<(int)T, thr256, 0, stream>>>(x, anw, h_b, (int)D);

    // 3) q/k/v projections (B = transposed weights, N x K)
    gemm_bf16_wmma<<<dim3(16, 32, 1), thr128, 0, stream>>>(
        h_b, wqT, qf, (int)T, (int)Qd, (int)D, (int)D, (int)D, (int)Qd,
        1, 1, 0, 0, 0, 0, 0, 0, nullptr, 0, nullptr, nullptr, nullptr, nullptr);
    gemm_bf16_wmma<<<dim3(4, 32, 1), thr128, 0, stream>>>(
        h_b, wkT, kf, (int)T, (int)KVd, (int)D, (int)D, (int)D, (int)KVd,
        1, 1, 0, 0, 0, 0, 0, 0, nullptr, 0, nullptr, nullptr, nullptr, nullptr);
    gemm_bf16_wmma<<<dim3(4, 32, 1), thr128, 0, stream>>>(
        h_b, wvT, vf, (int)T, (int)KVd, (int)D, (int)D, (int)D, (int)KVd,
        1, 1, 0, 0, 0, 0, 0, 0, nullptr, 0, nullptr, nullptr, nullptr, nullptr);

    // 4) RoPE(q,k) -> bf16 ; v -> bf16 transposed per batch: [b][kvh*hd][s]
    {
        i64 nq = T * H * 64, nk = T * KVH * 64;
        rope_cvt<<<(int)((nq + 255) / 256), thr256, 0, stream>>>(qf, q_b, fcos, fsin, (int)S, (int)H, nq);
        rope_cvt<<<(int)((nk + 255) / 256), thr256, 0, stream>>>(kf, k_b, fcos, fsin, (int)S, (int)KVH, nk);
        transpose_cvt_bf16<<<dim3(16, 32, (unsigned)Bn), thr256, 0, stream>>>(
            vf, vT_b, (int)S, (int)KVd, S * KVd, KVd * S);
    }

    // 5) scores[b,h] = q[b,:,h,:] @ k[b,:,h/4,:]^T  (k already N x K layout)
    gemm_bf16_wmma<<<dim3(8, 16, (unsigned)(Bn * H)), thr128, 0, stream>>>(
        q_b, k_b, scores, (int)S, (int)S, (int)HD, (int)Qd, (int)KVd, (int)S,
        (int)H, (int)(H / KVH),
        S * Qd, HD, S * KVd, HD, H * S * S, S * S,
        nullptr, 0, nullptr, nullptr, nullptr, nullptr);

    // 6) causal softmax -> bf16 probs
    softmax_causal<<<(int)(Bn * H * S), thr128, 0, stream>>>(scores, probs, (int)S, scale);

    // 7) attn[b,:,h,:] = probs[b,h] @ v[b,:,h/4,:]  (vT: rows=hd, cols=s)
    gemm_bf16_wmma<<<dim3(1, 16, (unsigned)(Bn * H)), thr128, 0, stream>>>(
        probs, vT_b, attn_f, (int)S, (int)HD, (int)S, (int)S, (int)S, (int)Qd,
        (int)H, (int)(H / KVH),
        H * S * S, S * S, KVd * S, (i64)HD * S, S * Qd, HD,
        nullptr, 0, nullptr, nullptr, nullptr, nullptr);

    cvt_f32_bf16<<<8192, thr256, 0, stream>>>(attn_f, attn_b, T * Qd);

    // 8) h1 = x + attn @ wo
    gemm_bf16_wmma<<<dim3(16, 32, 1), thr128, 0, stream>>>(
        attn_b, woT, h1, (int)T, (int)D, (int)Qd, (int)Qd, (int)Qd, (int)D,
        1, 1, 0, 0, 0, 0, 0, 0, x, (int)D, nullptr, nullptr, nullptr, nullptr);

    // 9) t = rmsnorm(h1, ffn_norm_w)
    rmsnorm_bf16<<<(int)T, thr256, 0, stream>>>(h1, fnw, t_b, (int)D);

    // 10) gate: top-2 routing lists
    zero_i32<<<1, 32, 0, stream>>>(counts, (int)E);
    gate_topk<<<(int)T, thr128, 0, stream>>>(t_b, gw, coefL, idxL, counts, (int)T, (int)D);

    // 11) out = h1 (residual base), then experts scatter-add coef*ff
    copy_f32<<<8192, thr256, 0, stream>>>(h1, out, T * D);

    for (int e = 0; e < (int)E; ++e) {
        const int* gl = idxL + (i64)e * T;
        const float* cl = coefL + (i64)e * T;
        const int* cnt = counts + e;
        // g = gather(t) @ w1[e]   (w1T: [HID][D])
        gemm_bf16_wmma<<<dim3(32, 32, 1), thr128, 0, stream>>>(
            t_b, w1T + (i64)e * HID * D, gbuf, (int)T, (int)HID, (int)D,
            (int)D, (int)D, (int)HID, 1, 1, 0, 0, 0, 0, 0, 0,
            nullptr, 0, gl, nullptr, nullptr, cnt);
        // u = gather(t) @ w3[e]
        gemm_bf16_wmma<<<dim3(32, 32, 1), thr128, 0, stream>>>(
            t_b, w3T + (i64)e * HID * D, ubuf, (int)T, (int)HID, (int)D,
            (int)D, (int)D, (int)HID, 1, 1, 0, 0, 0, 0, 0, 0,
            nullptr, 0, gl, nullptr, nullptr, cnt);
        // a = bf16(silu(g) * u)
        silu_mul_bf16<<<8192, thr256, 0, stream>>>(gbuf, ubuf, a_b, T * HID);
        // out[tok] += coef * (a @ w2[e])   (w2T: [D][HID])
        gemm_bf16_wmma<<<dim3(16, 32, 1), thr128, 0, stream>>>(
            a_b, w2T + (i64)e * HID * D, out, (int)T, (int)D, (int)HID,
            (int)HID, (int)HID, (int)D, 1, 1, 0, 0, 0, 0, 0, 0,
            nullptr, 0, nullptr, gl, cl, cnt);
    }
}
